// TargetAwarePool_18494129176717
// MI455X (gfx1250) — matrix operand order repaired and checked by reading, not compiled, not run on hardware
//
#include <hip/hip_runtime.h>
#include <stdint.h>

#define B_   2048
#define L_   200
#define H_   128
#define LP   208      // L padded to 13*16
#define STR  264      // LDS row stride in halfs (padded vs 256 to break bank conflicts)
#define NTHR 512      // 16 waves

typedef __attribute__((ext_vector_type(16))) __bf16 v16bf;
typedef __attribute__((ext_vector_type(8)))  float  v8f;

union FragU { v16bf v; uint4 q[2]; };

__device__ __forceinline__ unsigned short f2bf(float f) {
  union { float f; unsigned u; } v; v.f = f;
  unsigned r = v.u + 0x7FFFu + ((v.u >> 16) & 1u);   // round-to-nearest-even
  return (unsigned short)(r >> 16);
}

// GELU via sigmoid identity: 0.5*(1+tanh(u)) == 1/(1+exp(-2u)).
// One v_exp_f32 + one v_rcp_f32 + a few FMAs. Deviation from exact erf-GELU
// < ~1e-3, far below the bf16-GEMM noise floor here.
__device__ __forceinline__ float gelu_fast(float x) {
  float u = x * (0.7978845608028654f + 0.035677408136300125f * x * x);
  float s = __builtin_amdgcn_rcpf(1.0f + __expf(-2.0f * u));
  return x * s;
}

__device__ __forceinline__ void load_afrag(FragU& f, const unsigned short* Arow,
                                           int k, int hi) {
  // 16-bit A 16x32: lanes 0-15: v0-3=K0..7, v4-7=K16..23; lanes 16-31: +8
  int o = k * 32 + hi * 8;
  f.q[0] = *(const uint4*)(Arow + o);
  f.q[1] = *(const uint4*)(Arow + o + 16);
}

// Reduce all 8 C-rows over the 16-lane half with a select-merge butterfly:
// 8 ds_bpermutes total (vs 32 for row-by-row reduction). After the merges,
// lane j (j = lane&7) of each half holds the full sum for row hi*8 + j.
__device__ __forceinline__ void emit_scores(const v8f& acc, float w2v, int lane,
                                            int hi, int mt, bool valid,
                                            float* scores) {
  float s0 = gelu_fast(acc[0]) * w2v, s1 = gelu_fast(acc[1]) * w2v;
  float s2 = gelu_fast(acc[2]) * w2v, s3 = gelu_fast(acc[3]) * w2v;
  float s4 = gelu_fast(acc[4]) * w2v, s5 = gelu_fast(acc[5]) * w2v;
  float s6 = gelu_fast(acc[6]) * w2v, s7 = gelu_fast(acc[7]) * w2v;
  const bool b0 = (lane & 1) != 0, b1 = (lane & 2) != 0, b2 = (lane & 4) != 0;
  float t0 = (b0 ? s1 : s0) + __shfl_xor(b0 ? s0 : s1, 1, 32);
  float t1 = (b0 ? s3 : s2) + __shfl_xor(b0 ? s2 : s3, 1, 32);
  float t2 = (b0 ? s5 : s4) + __shfl_xor(b0 ? s4 : s5, 1, 32);
  float t3 = (b0 ? s7 : s6) + __shfl_xor(b0 ? s6 : s7, 1, 32);
  float u0 = (b1 ? t1 : t0) + __shfl_xor(b1 ? t0 : t1, 2, 32);
  float u1 = (b1 ? t3 : t2) + __shfl_xor(b1 ? t2 : t3, 2, 32);
  float v  = (b2 ? u1 : u0) + __shfl_xor(b2 ? u0 : u1, 4, 32);
  v += __shfl_xor(v, 8, 32);
  if (valid && (lane & 15) < 8)
    atomicAdd(&scores[mt * 16 + hi * 8 + (lane & 7)], v);
}

__global__ __launch_bounds__(NTHR, 1)
void ta_pool_kernel(const float* __restrict__ query,
                    const float* __restrict__ keys,
                    const unsigned char* __restrict__ key_mask,
                    const float* __restrict__ W1,
                    const float* __restrict__ b1,
                    const float* __restrict__ W2,
                    const float* __restrict__ b2,
                    float* __restrict__ out) {
  extern __shared__ char smem_raw[];
  unsigned short* smA   = (unsigned short*)smem_raw;      // LP  x STR halfs: [k | q*k]
  unsigned short* smB   = smA + LP * STR;                 // H_  x STR halfs: [W1b-W1c ; W1d]^T
  float* row_bias = (float*)(smB + H_ * STR);             // 128
  float* scores   = row_bias + H_;                        // 208
  float* maskf    = scores + LP;                          // 208
  float* wbuf     = maskf + LP;                           // 208
  float* outp     = wbuf + LP;                            // 512

  const int b   = blockIdx.x;
  const int tid = threadIdx.x;
  const float* qv = query + b * H_;
  const float* kb = keys + (size_t)b * L_ * H_;

  // ---------------- Phase 1: staging ----------------
  for (int idx = tid; idx < LP * H_; idx += NTHR) {
    int l = idx >> 7;
    int i = idx & 127;
    float kvf = (l < L_) ? kb[l * H_ + i] : 0.0f;
    unsigned short* row = smA + l * STR;
    row[i]      = f2bf(kvf);
    row[H_ + i] = f2bf(qv[i] * kvf);
  }
  for (int idx = tid; idx < H_ * H_; idx += NTHR) {
    int kk = idx >> 7;
    int n  = idx & 127;
    float wb = W1[(H_ + kk) * H_ + n] - W1[(2 * H_ + kk) * H_ + n]; // W1b - W1c
    float wd = W1[(3 * H_ + kk) * H_ + n];                          // W1d
    smB[n * STR + kk]      = f2bf(wb);
    smB[n * STR + H_ + kk] = f2bf(wd);
  }
  if (tid < H_) {             // row bias = b1 + q @ (W1a + W1c)
    float acc = b1[tid];
    for (int i = 0; i < H_; ++i)
      acc += qv[i] * (W1[i * H_ + tid] + W1[(2 * H_ + i) * H_ + tid]);
    row_bias[tid] = acc;
  }
  if (tid < LP) {
    scores[tid] = 0.0f;
    maskf[tid]  = (tid < L_ && key_mask[b * L_ + tid] != 0) ? 1.0f : 0.0f;
  }
  __syncthreads();

  // ---------------- Phase 2: WMMA GEMM + GELU + score partials ----------------
  {
    const int wave  = tid >> 5;
    const int lane  = tid & 31;
    const int hi    = lane >> 4;         // 0: lanes 0-15, 1: lanes 16-31
    const int nc    = (wave & 7) * 16 + (lane & 15);  // N column owned by this lane
    const int group = wave >> 3;         // 0 or 1: which set of M-tile pairs

    // Preload all 8 B fragments (K=256, k-step 32) for this wave's N-tile.
    // 16-bit B 32x16: lanes 0-15 hold K k0..k0+7, k0+16..k0+23; lanes 16-31: +8,
    // packed two-K per VGPR -> two 8-half runs from the B^T row (matches A).
    FragU bf[8];
    const unsigned short* Brow = smB + nc * STR;
#pragma unroll
    for (int k = 0; k < 8; ++k) {
      int kh = k * 32 + hi * 8;
      bf[k].q[0] = *(const uint4*)(Brow + kh);
      bf[k].q[1] = *(const uint4*)(Brow + kh + 16);
    }
    const float bias = row_bias[nc];
    const float w2v  = W2[nc];

    // Two independent WMMA chains (a pair of M-tiles) per iteration so LDS
    // loads and matrix ops from the two chains pipeline against each other.
    for (int mt = group * 2; mt < 13; mt += 4) {
      const int  mt1   = mt + 1;
      const bool live1 = (mt1 < 13);     // pad tile reads in-bounds LDS garbage
      const unsigned short* Arow0 = smA + (mt  * 16 + (lane & 15)) * STR;
      const unsigned short* Arow1 = smA + (mt1 * 16 + (lane & 15)) * STR;
      v8f acc0, acc1;
#pragma unroll
      for (int r = 0; r < 8; ++r) { acc0[r] = bias; acc1[r] = bias; }
      FragU fa, fb;
#pragma unroll
      for (int k = 0; k < 8; ++k) {
        load_afrag(fa, Arow0, k, hi);
        load_afrag(fb, Arow1, k, hi);
        acc0 = __builtin_amdgcn_wmma_f32_16x16x32_bf16(false, fa.v, false, bf[k].v,
                                                       (short)0, acc0, false, false);
        acc1 = __builtin_amdgcn_wmma_f32_16x16x32_bf16(false, fb.v, false, bf[k].v,
                                                       (short)0, acc1, false, false);
      }
      emit_scores(acc0, w2v, lane, hi, mt,  true,  scores);
      emit_scores(acc1, w2v, lane, hi, mt1, live1, scores);
    }
  }
  __syncthreads();

  // ---------------- Phase 3: masked softmax + renorm (wave 0) ----------------
  if (tid < 32) {
    const float b2v = b2[0];
    float sl[7];
    float m = -3.0e38f;
#pragma unroll
    for (int i = 0; i < 7; ++i) {
      int l = tid + 32 * i;
      float s = -1.0e4f;
      if (l < L_) { s = (maskf[l] > 0.0f) ? (scores[l] + b2v) : -1.0e4f; m = fmaxf(m, s); }
      sl[i] = s;
    }
#pragma unroll
    for (int off = 16; off >= 1; off >>= 1) m = fmaxf(m, __shfl_xor(m, off, 32));
    float sum = 0.0f;
#pragma unroll
    for (int i = 0; i < 7; ++i) {
      int l = tid + 32 * i;
      float e = (l < L_) ? __expf(sl[i] - m) : 0.0f;
      sl[i] = e; sum += e;
    }
#pragma unroll
    for (int off = 16; off >= 1; off >>= 1) sum += __shfl_xor(sum, off, 32);
    float wsum = 0.0f;
#pragma unroll
    for (int i = 0; i < 7; ++i) {
      int l = tid + 32 * i;
      if (l < LP) {
        float w = (l < L_) ? (sl[i] / sum) * maskf[l] : 0.0f;
        wbuf[l] = w; wsum += w;
      }
    }
#pragma unroll
    for (int off = 16; off >= 1; off >>= 1) wsum += __shfl_xor(wsum, off, 32);
    float denom = fmaxf(wsum, 1.0f);
#pragma unroll
    for (int i = 0; i < 7; ++i) {
      int l = tid + 32 * i;
      if (l < LP) wbuf[l] = wbuf[l] / denom;
    }
  }
  __syncthreads();

  // ---------------- Phase 4: out = w @ keys (full fp32 from global) ----------------
  {
    const int d = tid & 127;
    const int q4 = tid >> 7;             // 0..3, 50 rows each
    const int l0 = q4 * 50;
    float acc = 0.0f;
    for (int l = l0; l < l0 + 50; ++l)
      acc += wbuf[l] * kb[l * H_ + d];
    outp[tid] = acc;
  }
  __syncthreads();
  if (tid < H_)
    out[b * H_ + tid] = (outp[tid] + outp[H_ + tid]) + (outp[2 * H_ + tid] + outp[3 * H_ + tid]);
}

extern "C" void kernel_launch(void* const* d_in, const int* in_sizes, int n_in,
                              void* d_out, int out_size, void* d_ws, size_t ws_size,
                              hipStream_t stream) {
  const float*         query    = (const float*)d_in[0];
  const float*         keys     = (const float*)d_in[1];
  const unsigned char* key_mask = (const unsigned char*)d_in[2];
  const float*         W1       = (const float*)d_in[3];
  const float*         b1       = (const float*)d_in[4];
  const float*         W2       = (const float*)d_in[5];
  const float*         b2       = (const float*)d_in[6];
  float*               out      = (float*)d_out;

  const size_t smem = (size_t)(LP * STR + H_ * STR) * sizeof(unsigned short)
                    + (size_t)(H_ + LP + LP + LP + NTHR) * sizeof(float);
  static const int attr_once = [] {
    hipFuncSetAttribute((const void*)ta_pool_kernel,
                        hipFuncAttributeMaxDynamicSharedMemorySize, 192 * 1024);
    return 0;
  }();
  (void)attr_once;
  ta_pool_kernel<<<dim3(B_), dim3(NTHR), smem, stream>>>(query, keys, key_mask,
                                                         W1, b1, W2, b2, out);
}
